// TemporalAttention_67242007986635
// MI455X (gfx1250) — matrix operand order repaired and checked by reading, not compiled, not run on hardware
//
#include <hip/hip_runtime.h>

// ---------------------------------------------------------------------------
// Types for CDNA5 WMMA (gfx1250, wave32)
// ---------------------------------------------------------------------------
typedef __bf16 bf16;
typedef bf16         v16bf __attribute__((ext_vector_type(16)));
typedef float        v8f   __attribute__((ext_vector_type(8)));
typedef unsigned int v4u   __attribute__((ext_vector_type(4)));

union Frag16 { v4u q[2]; v16bf b; };

// Problem constants (from reference)
#define BATCH_TOT 16384
#define NNEI      20
#define OUTD      272      // 17 tiles of 16
#define KEYD      444      // pad -> 448 = 14 ksteps of 32
#define KEYD_P    448
#define OUTD_P    288      // pad -> 9 ksteps of 32
#define NTILES    17
#define KS_KV     14
#define KS_O      9
#define BPW       4        // batch items per workgroup
#define ROWS      (BPW*NNEI)   // 80 rows = 5 mtiles
#define ASTR_KV   232      // LDS A stride (halves), 464B -> conflict-free-ish
#define ASTR_O    296      // LDS A stride for 288-col A
#define FRAG_H    512      // halves per packed B fragment (32 lanes * 16)

#define SK_HALF   (NTILES*KS_KV*FRAG_H)   // 121856 per kv weight
#define SQ_HALF   (NTILES*KS_O*FRAG_H)    // 78336 per out weight
#define PACK_TOT  (2*SK_HALF + 2*SQ_HALF) // 400384 halves = 800768 B in d_ws

__device__ __forceinline__ unsigned short f2bf(float f) {
  unsigned u = __float_as_uint(f);
  u += 0x7fffu + ((u >> 16) & 1u);          // round-to-nearest-even
  return (unsigned short)(u >> 16);
}
__device__ __forceinline__ float bf2f(unsigned short h) {
  return __uint_as_float(((unsigned)h) << 16);
}

// A fragment (16x32 bf16) from row-major LDS tile. ISA 7.12.2 16-bit A layout:
// lane L: M = L%16, hi = L/16; halves j<8 -> K=kbase+8*hi+j ; j>=8 -> K=kbase+16+8*hi+(j-8)
__device__ __forceinline__ v16bf load_afrag(const unsigned short* As, int stride,
                                            int kbase, int lane) {
  const int m = lane & 15, hi = lane >> 4;
  Frag16 f;
  f.q[0] = *(const v4u*)(As + m * stride + kbase + 8 * hi);
  f.q[1] = *(const v4u*)(As + m * stride + kbase + 16 + 8 * hi);
  return f.b;
}
// B fragment (32x16 bf16) pre-packed in fragment order: 32B contiguous per lane.
__device__ __forceinline__ v16bf load_bfrag(const unsigned short* g, int lane) {
  const v4u* p = (const v4u*)(g + lane * 16);
  Frag16 f;
  f.q[0] = p[0];
  f.q[1] = p[1];
  return f.b;
}
__device__ __forceinline__ v8f wmma_bf16(v16bf a, v16bf b, v8f c) {
  return __builtin_amdgcn_wmma_f32_16x16x32_bf16(false, a, false, b, (short)0, c,
                                                 false, false);
}

// ---------------------------------------------------------------------------
// Kernel 1: pack W (torch layout [out,in], B = W^T) into WMMA B-fragment order.
// dst[(nt*KS + ks)*512 + lane*16 + j] = bf16(W[n][k]),
//   n = nt*16 + (lane&15), k = ks*32 + 8*(lane>>4) + j + (j>=8 ? 8 : 0); zero-pad k >= KIN.
// ---------------------------------------------------------------------------
__global__ void pack_weights(const float* __restrict__ Wq,
                             const float* __restrict__ Wk,
                             const float* __restrict__ Wv,
                             const float* __restrict__ Wr,
                             unsigned short* __restrict__ ws) {
  int idx = blockIdx.x * 256 + threadIdx.x;
  if (idx >= PACK_TOT) return;
  const float* W; unsigned short* dst; int KS, KIN, local;
  if (idx < SK_HALF)                { W = Wk; dst = ws;                    KS = KS_KV; KIN = KEYD; local = idx; }
  else if (idx < 2*SK_HALF)         { W = Wv; dst = ws + SK_HALF;          KS = KS_KV; KIN = KEYD; local = idx - SK_HALF; }
  else if (idx < 2*SK_HALF+SQ_HALF) { W = Wq; dst = ws + 2*SK_HALF;        KS = KS_O;  KIN = OUTD; local = idx - 2*SK_HALF; }
  else                              { W = Wr; dst = ws + 2*SK_HALF+SQ_HALF;KS = KS_O;  KIN = OUTD; local = idx - 2*SK_HALF - SQ_HALF; }
  int frag = local >> 9, within = local & 511;
  int lane = within >> 4, j = within & 15;
  int nt = frag / KS, ks = frag % KS;
  int n = nt * 16 + (lane & 15);
  int k = ks * 32 + 8 * (lane >> 4) + j + ((j >= 8) ? 8 : 0);
  float v = (k < KIN) ? W[n * KIN + k] : 0.f;
  dst[local] = f2bf(v);
}

// ---------------------------------------------------------------------------
// Kernel 2: fully fused temporal attention. 1 block = 4 batch items.
// ---------------------------------------------------------------------------
struct __align__(16) Smem {
  unsigned short As[16 * ASTR_O];   // 9472 B  (A staging, reused by all GEMMs)
  unsigned short V[ROWS * OUTD];    // 43520 B (V tile, bf16)
  float qp[BPW * OUTD];             // 4352 B  (projected q)
  float attn[BPW * 4 * NNEI];       // 1280 B  (score partials -> scores)
  float xo[BPW * OUTD];             // 4352 B  (attn output -> final x)
  float stats[2 * BPW + 2];         // LN mean / rsigma
};                                   // total 63 KB < 64 KB static LDS

__global__ __launch_bounds__(256)
void fused_attn(const float* __restrict__ nodef,   // [B,172]
                const float* __restrict__ ntimef,  // [B,1,100]
                const float* __restrict__ nbf,     // [B,N,172]
                const float* __restrict__ nbtf,    // [B,N,100]
                const float* __restrict__ nbef,    // [B,N,172]
                const int*   __restrict__ nmask,   // [B,N]
                const float* __restrict__ br,
                const float* __restrict__ gammap,
                const float* __restrict__ betap,
                const unsigned short* __restrict__ WkP,
                const unsigned short* __restrict__ WvP,
                const unsigned short* __restrict__ WqP,
                const unsigned short* __restrict__ WrP,
                float* __restrict__ outp) {
  __shared__ Smem s;
  const int tid  = threadIdx.x;
  const int lane = tid & 31;
  // readfirstlane -> SGPR: guarantees scalar branches around WMMA (EXEC stays all-1)
  const int wave = __builtin_amdgcn_readfirstlane(tid >> 5);
  const long long b0 = (long long)blockIdx.x * BPW;
  const int nt0 = wave, nt1 = wave + 8;   // each wave owns ntiles {w, w+8}, wave0 also 16
  const int ln = lane & 15, hi = lane >> 4;

  for (int i = tid; i < BPW * 4 * NNEI; i += 256) s.attn[i] = 0.f;

  // ---------------- Phase Q: q = q_in @ Wq^T  (1 mtile, rows 0..3 valid) ----
  for (int idx = tid; idx < 16 * OUTD_P; idx += 256) {
    int r = idx / OUTD_P, c = idx % OUTD_P;
    float v = 0.f;
    if (r < BPW && c < OUTD) {
      long long bg = b0 + r;
      v = (c < 172) ? nodef[bg * 172 + c] : ntimef[bg * 100 + (c - 172)];
    }
    s.As[r * ASTR_O + c] = f2bf(v);
  }
  __syncthreads();
  {
    v8f c0 = {}, c1 = {}, c2 = {};
    for (int ks = 0; ks < KS_O; ++ks) {
      v16bf a = load_afrag(s.As, ASTR_O, ks * 32, lane);
      c0 = wmma_bf16(a, load_bfrag(WqP + (nt0 * KS_O + ks) * FRAG_H, lane), c0);
      c1 = wmma_bf16(a, load_bfrag(WqP + (nt1 * KS_O + ks) * FRAG_H, lane), c1);
      if (wave == 0)
        c2 = wmma_bf16(a, load_bfrag(WqP + (16 * KS_O + ks) * FRAG_H, lane), c2);
    }
    if (hi == 0) {                       // rows 0..3 live in lanes 0-15, regs 0-3
      #pragma unroll
      for (int r = 0; r < BPW; ++r) {
        s.qp[r * OUTD + nt0 * 16 + ln] = c0[r];
        s.qp[r * OUTD + nt1 * 16 + ln] = c1[r];
        if (wave == 0) s.qp[r * OUTD + 256 + ln] = c2[r];
      }
    }
  }
  __syncthreads();

  // ---------------- Phase KV: K,V = kv_in @ Wk^T / Wv^T over 5 mtiles -------
  for (int mt = 0; mt < 5; ++mt) {
    v8f k0 = {}, k1 = {}, k2 = {}, v0 = {}, v1 = {}, v2 = {};
    for (int ch = 0; ch < 2; ++ch) {            // stream A in 2 chunks of 224 cols
      __syncthreads();
      for (int idx = tid; idx < 16 * 224; idx += 256) {
        int r = idx / 224, cc = idx % 224, c = ch * 224 + cc;
        int row = mt * 16 + r;
        long long bn = (b0 + row / NNEI) * NNEI + (row % NNEI);
        float v = 0.f;
        if (c < 172)      v = nbf [bn * 172 + c];
        else if (c < 344) v = nbef[bn * 172 + (c - 172)];
        else if (c < 444) v = nbtf[bn * 100 + (c - 344)];
        s.As[r * ASTR_KV + cc] = f2bf(v);
      }
      __syncthreads();
      for (int ks = 0; ks < 7; ++ks) {
        int ksg = ch * 7 + ks;
        const unsigned short* pk0 = WkP + (nt0 * KS_KV + ksg) * FRAG_H;
        const unsigned short* pv0 = WvP + (nt0 * KS_KV + ksg) * FRAG_H;
        __builtin_prefetch(pk0 + FRAG_H, 0, 1);  // next-kstep weight stream
        v16bf a = load_afrag(s.As, ASTR_KV, ks * 32, lane);
        k0 = wmma_bf16(a, load_bfrag(pk0, lane), k0);
        v0 = wmma_bf16(a, load_bfrag(pv0, lane), v0);
        k1 = wmma_bf16(a, load_bfrag(WkP + (nt1 * KS_KV + ksg) * FRAG_H, lane), k1);
        v1 = wmma_bf16(a, load_bfrag(WvP + (nt1 * KS_KV + ksg) * FRAG_H, lane), v1);
        if (wave == 0) {
          k2 = wmma_bf16(a, load_bfrag(WkP + (16 * KS_KV + ksg) * FRAG_H, lane), k2);
          v2 = wmma_bf16(a, load_bfrag(WvP + (16 * KS_KV + ksg) * FRAG_H, lane), v2);
        }
      }
    }
    // Emit: K frag -> score partials (q . k via LDS atomics); V frag -> LDS bf16
    #pragma unroll
    for (int r = 0; r < 8; ++r) {
      int row = mt * 16 + r + 8 * hi;           // 0..79
      int bl = row / NNEI, n = row % NNEI;
      {
        int col = nt0 * 16 + ln;
        atomicAdd(&s.attn[(bl * 4 + col / 68) * NNEI + n], k0[r] * s.qp[bl * OUTD + col]);
        s.V[row * OUTD + col] = f2bf(v0[r]);
      }
      {
        int col = nt1 * 16 + ln;
        atomicAdd(&s.attn[(bl * 4 + col / 68) * NNEI + n], k1[r] * s.qp[bl * OUTD + col]);
        s.V[row * OUTD + col] = f2bf(v1[r]);
      }
      if (wave == 0) {
        int col = 256 + ln;
        atomicAdd(&s.attn[(bl * 4 + col / 68) * NNEI + n], k2[r] * s.qp[bl * OUTD + col]);
        s.V[row * OUTD + col] = f2bf(v2[r]);
      }
    }
  }
  __syncthreads();

  // ---------------- Masked softmax over neighbors (16 (b,h) pairs) ----------
  const float scale = 0.121267813f;   // 68^-0.5
  if (tid < 16) {
    int bl = tid >> 2, h = tid & 3;
    long long bg = b0 + bl;
    float vb[NNEI], mx = -3.4e38f;
    for (int n = 0; n < NNEI; ++n) {
      float v = (nmask[bg * NNEI + n] != 0) ? s.attn[(bl * 4 + h) * NNEI + n] * scale
                                            : -1e10f;
      vb[n] = v; mx = fmaxf(mx, v);
    }
    float sum = 0.f;
    for (int n = 0; n < NNEI; ++n) { float e = __expf(vb[n] - mx); vb[n] = e; sum += e; }
    float inv = 1.f / sum;
    for (int n = 0; n < NNEI; ++n) s.attn[(bl * 4 + h) * NNEI + n] = vb[n] * inv;
  }
  __syncthreads();

  // ---------------- out = scores @ V -----------------------------------------
  for (int idx = tid; idx < BPW * OUTD; idx += 256) {
    int bl = idx / OUTD, col = idx % OUTD;
    const float* sc = &s.attn[(bl * 4 + col / 68) * NNEI];
    const unsigned short* vrow = &s.V[(bl * NNEI) * OUTD + col];
    float acc = 0.f;
    #pragma unroll
    for (int n = 0; n < NNEI; ++n) acc += sc[n] * bf2f(vrow[n * OUTD]);
    s.xo[idx] = acc;
  }
  __syncthreads();

  // ---------------- x = out @ Wr^T + br + residual ---------------------------
  for (int idx = tid; idx < 16 * OUTD_P; idx += 256) {
    int r = idx / OUTD_P, c = idx % OUTD_P;
    float v = (r < BPW && c < OUTD) ? s.xo[r * OUTD + c] : 0.f;
    s.As[r * ASTR_O + c] = f2bf(v);
  }
  __syncthreads();
  {
    v8f c0 = {}, c1 = {}, c2 = {};
    for (int ks = 0; ks < KS_O; ++ks) {
      v16bf a = load_afrag(s.As, ASTR_O, ks * 32, lane);
      c0 = wmma_bf16(a, load_bfrag(WrP + (nt0 * KS_O + ks) * FRAG_H, lane), c0);
      c1 = wmma_bf16(a, load_bfrag(WrP + (nt1 * KS_O + ks) * FRAG_H, lane), c1);
      if (wave == 0)
        c2 = wmma_bf16(a, load_bfrag(WrP + (16 * KS_O + ks) * FRAG_H, lane), c2);
    }
    if (hi == 0) {
      #pragma unroll
      for (int r = 0; r < BPW; ++r) {
        long long bg = b0 + r;
        {
          int col = nt0 * 16 + ln;
          float res = (col < 172) ? nodef[bg * 172 + col] : ntimef[bg * 100 + col - 172];
          s.xo[r * OUTD + col] = c0[r] + br[col] + res;
        }
        {
          int col = nt1 * 16 + ln;
          float res = (col < 172) ? nodef[bg * 172 + col] : ntimef[bg * 100 + col - 172];
          s.xo[r * OUTD + col] = c1[r] + br[col] + res;
        }
        if (wave == 0) {
          int col = 256 + ln;
          float res = (col < 172) ? nodef[bg * 172 + col] : ntimef[bg * 100 + col - 172];
          s.xo[r * OUTD + col] = c2[r] + br[col] + res;
        }
      }
    }
  }
  __syncthreads();

  // ---------------- LayerNorm ------------------------------------------------
  if (wave < BPW) {
    float sm = 0.f, sq = 0.f;
    for (int c = lane; c < OUTD; c += 32) {
      float x = s.xo[wave * OUTD + c];
      sm += x; sq += x * x;
    }
    for (int off = 16; off > 0; off >>= 1) {
      sm += __shfl_xor(sm, off, 32);
      sq += __shfl_xor(sq, off, 32);
    }
    if (lane == 0) {
      float mu = sm * (1.f / OUTD);
      float var = sq * (1.f / OUTD) - mu * mu;
      s.stats[wave * 2]     = mu;
      s.stats[wave * 2 + 1] = rsqrtf(var + 1e-5f);
    }
  }
  __syncthreads();
  for (int idx = tid; idx < BPW * OUTD; idx += 256) {
    int bl = idx / OUTD, c = idx % OUTD;
    float y = (s.xo[idx] - s.stats[bl * 2]) * s.stats[bl * 2 + 1];
    outp[(b0 + bl) * OUTD + c] = y * gammap[c] + betap[c];
  }
}

// ---------------------------------------------------------------------------
extern "C" void kernel_launch(void* const* d_in, const int* in_sizes, int n_in,
                              void* d_out, int out_size, void* d_ws, size_t ws_size,
                              hipStream_t stream) {
  const float* nodef  = (const float*)d_in[0];
  const float* ntimef = (const float*)d_in[1];
  const float* nbf    = (const float*)d_in[2];
  const float* nbtf   = (const float*)d_in[3];
  const float* nbef   = (const float*)d_in[4];
  const int*   nmask  = (const int*)  d_in[5];
  const float* Wq     = (const float*)d_in[6];
  const float* Wk     = (const float*)d_in[7];
  const float* Wv     = (const float*)d_in[8];
  const float* Wr     = (const float*)d_in[9];
  const float* br     = (const float*)d_in[10];
  const float* gammap = (const float*)d_in[11];
  const float* betap  = (const float*)d_in[12];

  unsigned short* wsu = (unsigned short*)d_ws;
  const int B = in_sizes[0] / 172;      // 16384

  pack_weights<<<(PACK_TOT + 255) / 256, 256, 0, stream>>>(Wq, Wk, Wv, Wr, wsu);

  const unsigned short* WkP = wsu;
  const unsigned short* WvP = wsu + SK_HALF;
  const unsigned short* WqP = wsu + 2 * SK_HALF;
  const unsigned short* WrP = wsu + 2 * SK_HALF + SQ_HALF;

  fused_attn<<<B / BPW, 256, 0, stream>>>(nodef, ntimef, nbf, nbtf, nbef, nmask,
                                          br, gammap, betap,
                                          WkP, WvP, WqP, WrP, (float*)d_out);
}
